// GNN_14465449853013
// MI455X (gfx1250) — compile-verified, matrix operand order checked by reading, not run on hardware
//
#include <hip/hip_runtime.h>
#include <hip/hip_bf16.h>

// ---------------------------------------------------------------------------
// 2-layer GCN on MI455X (gfx1250, wave32, WMMA + TDM).
// Dominant op: xw1 = x[16384x16384] @ W1[16384x50]  (bf16 WMMA, f32 accum)
// A-tiles (64 rows x 128 f32) staged global->LDS by the Tensor Data Mover,
// double-buffered (TENSORcnt) so DMA of chunk i+1 overlaps compute of chunk i.
// Each wave computes TWO 16x16 output tiles sharing one A fragment: the two
// independent WMMA chains fill each other's post-WMMA hazard slots.
// ---------------------------------------------------------------------------

typedef __attribute__((ext_vector_type(16))) __bf16    v16bf;
typedef __attribute__((ext_vector_type(8)))  float     v8f;
typedef __attribute__((ext_vector_type(4)))  unsigned  u32x4;
typedef __attribute__((ext_vector_type(8)))  int       i32x8;
typedef __attribute__((ext_vector_type(4)))  int       i32x4;

#define NN    16384      // nodes
#define KD    16384      // feature dim (= K of big GEMM)
#define F1    50         // hidden dim layer 1
#define F1P   64         // padded to 4 WMMA col-tiles
#define F2    2          // hidden dim layer 2
#define NCLS  7          // classes
#define NKT   (KD / 32)  // 512 WMMA K-steps
#define KC    128        // K per TDM chunk (4 WMMA steps)
#define NCH   (KD / KC)  // 128 chunks
#define MROW  64         // rows per block
#define ROWP  132        // LDS row stride in dwords (128 + 4 TDM pad)
#define ABUF  (MROW * ROWP)       // dwords per A buffer
#define ABUFB (ABUF * 4)          // bytes per A buffer

// truncating f32 pair -> packed bf16 dword (1 VALU op)
__device__ __forceinline__ unsigned trunc2bf(float even, float odd) {
  return __builtin_amdgcn_perm(__float_as_uint(odd), __float_as_uint(even),
                               0x07060302u);
}

// round-to-nearest-even f32 -> bf16 pack (used once, for W1)
__device__ __forceinline__ unsigned pack_bf16(float lo, float hi) {
  unsigned a = __float_as_uint(lo);
  unsigned b = __float_as_uint(hi);
  a = (a + 0x7FFFu + ((a >> 16) & 1u)) >> 16;
  b = (b + 0x7FFFu + ((b >> 16) & 1u)) >> 16;
  return (a & 0xFFFFu) | (b << 16);
}

// ---- TDM: 2D tile (64 rows x 128 f32, row stride 16384 f32) -> LDS --------
// D# group0: count=1 | lds_addr | global_addr | type=2
// D# group1: data_size=4B, pad_enable, pad_interval=128dw, pad_amount=4dw,
//            tensor_dim0=16384, tensor_dim1=16384, tile_dim0=128, tile_dim1=64,
//            tensor_dim0_stride=16384
__device__ __forceinline__ void tdm_load_tile(const float* gsrc, unsigned lds_off) {
  unsigned long long ga = (unsigned long long)(uintptr_t)gsrc;
  u32x4 g0 = { 1u,
               lds_off,
               (unsigned)ga,
               (unsigned)((ga >> 32) & 0x01FFFFFFu) | (2u << 30) };
  i32x8 g1 = { (int)0x07920000,        // data_size=4B | pad_en | intv=128 | amt=4
               (int)0x40000000,        // tensor_dim0 low16 (=16384) in [31:16]
               (int)0x40000000,        // tensor_dim1 low16 (=16384) in [31:16]
               (int)(128u << 16),      // tile_dim0 = 128
               MROW,                   // tile_dim1 = 64 (tile_dim2 = 0)
               16384,                  // tensor_dim0_stride low32
               0, 0 };
  i32x4 z4 = {0, 0, 0, 0};
#if __has_include(<hip/amd_detail/amd_gfx1250_TDM.h>)
  i32x8 z8 = {0, 0, 0, 0, 0, 0, 0, 0};
  __builtin_amdgcn_tensor_load_to_lds(g0, g1, z4, z4, z8, 0);
#else
  __builtin_amdgcn_tensor_load_to_lds(g0, g1, z4, z4, 0);
#endif
}

// ---------------- degree / normalization ----------------------------------
__global__ void gcn_deg_init(float* deg) {
  int i = blockIdx.x * blockDim.x + threadIdx.x;
  if (i < NN) deg[i] = 1.0f;  // self loop
}

__global__ void gcn_deg_edges(const int* __restrict__ ei, int E, float* deg) {
  int e = blockIdx.x * blockDim.x + threadIdx.x;
  if (e < E) atomicAdd(&deg[ei[E + e]], 1.0f);  // dst row of edge_index
}

__global__ void gcn_dinv(const float* __restrict__ deg, float* dinv) {
  int i = blockIdx.x * blockDim.x + threadIdx.x;
  if (i < NN) dinv[i] = rsqrtf(deg[i]);
}

// ---------------- pack W1 into per-lane WMMA B-register order --------------
// B fragment (16-bit, 32x16): lane n holds col = n%16;
// lanes 0-15: K = {0..7,16..23}; lanes 16-31: K = {8..15,24..31};
// each of 8 dwords packs 2 consecutive K values.
__global__ void gcn_pack_w1(const float* __restrict__ W1, unsigned* __restrict__ Bpk) {
  int tid = blockIdx.x * blockDim.x + threadIdx.x;   // 0 .. NKT*4*32-1
  if (tid >= NKT * 4 * 32) return;
  int lane = tid & 31;
  int ct   = (tid >> 5) & 3;
  int kt   = tid >> 7;
  int n    = ct * 16 + (lane & 15);
  int base = (lane < 16) ? 0 : 8;
  unsigned* dst = Bpk + (size_t)tid * 8;
#pragma unroll
  for (int vg = 0; vg < 8; ++vg) {
    int koff = (vg < 4) ? (base + vg * 2) : (16 + base + (vg - 4) * 2);
    int k = kt * 32 + koff;
    float lo = (n < F1) ? W1[(size_t)k * F1 + n]       : 0.0f;
    float hi = (n < F1) ? W1[(size_t)(k + 1) * F1 + n] : 0.0f;
    dst[vg] = pack_bf16(lo, hi);
  }
}

// ---------------- big GEMM: xw1 = x @ W1 -----------------------------------
// grid: 256 blocks (64 rows each), 256 threads = 8 waves.
// wave w: M-tile ml = w>>1 (0..3); N-tiles {2*(w&1), 2*(w&1)+1} share one A.
__global__ __launch_bounds__(256) void gcn_gemm1(const float* __restrict__ x,
                                                 const unsigned* __restrict__ Bpk,
                                                 float* __restrict__ xw1) {
  extern __shared__ float ldsA[];          // 2 x (64 rows x 132 dwords)

  const int blk  = blockIdx.x;             // 0..255
  const int wave = threadIdx.x >> 5;
  const int lane = threadIdx.x & 31;
  const int ml   = wave >> 1;              // M tile within block (0..3)
  const int ct0  = (wave & 1) * 2;         // first of two N tiles
  const int row  = ml * 16 + (lane & 15);  // A row in LDS tile
  const int koff = (lane < 16) ? 0 : 8;    // A K-pattern base (f32 index)

  const unsigned lds_base = (unsigned)(uintptr_t)&ldsA[0];
  const float* tile0 = x + (size_t)blk * MROW * KD;

  v8f acc0 = {};
  v8f acc1 = {};

  if (threadIdx.x < 32) tdm_load_tile(tile0, lds_base);   // prime buffer 0

  for (int c = 0; c < NCH; ++c) {
    if (threadIdx.x < 32) {
      if (c + 1 < NCH) {
        tdm_load_tile(tile0 + (size_t)(c + 1) * KC,
                      lds_base + (unsigned)(((c + 1) & 1) * ABUFB));
        __builtin_amdgcn_s_wait_tensorcnt(1);   // chunk c landed
      } else {
        __builtin_amdgcn_s_wait_tensorcnt(0);
      }
    }
    __syncthreads();

    const float* abuf = &ldsA[(c & 1) * ABUF];
#pragma unroll
    for (int s = 0; s < 4; ++s) {
      const float* ap = abuf + row * ROWP + s * 32 + koff;
      union { unsigned u[8]; v16bf v; } A, B0, B1;
#pragma unroll
      for (int j = 0; j < 4; ++j)
        A.u[j] = trunc2bf(ap[2 * j], ap[2 * j + 1]);
#pragma unroll
      for (int j = 0; j < 4; ++j)
        A.u[4 + j] = trunc2bf(ap[16 + 2 * j], ap[16 + 2 * j + 1]);

      const unsigned* bp = Bpk + ((size_t)((c * 4 + s) * 4 + ct0) * 32 + lane) * 8;
#pragma unroll
      for (int j = 0; j < 8; ++j) B0.u[j] = bp[j];
#pragma unroll
      for (int j = 0; j < 8; ++j) B1.u[j] = bp[256 + j];   // next N tile (+32 lanes*8)

      acc0 = __builtin_amdgcn_wmma_f32_16x16x32_bf16(false, A.v, false, B0.v,
                                                     (short)0, acc0, false, false);
      acc1 = __builtin_amdgcn_wmma_f32_16x16x32_bf16(false, A.v, false, B1.v,
                                                     (short)0, acc1, false, false);
    }
    __syncthreads();
  }

  // C/D layout: VGPR j -> M = j (lanes 0-15) / 8+j (lanes 16-31), N = lane%16
  const int n0    = ct0 * 16 + (lane & 15);
  const int mbase = blk * MROW + ml * 16 + ((lane < 16) ? 0 : 8);
#pragma unroll
  for (int j = 0; j < 8; ++j) {
    xw1[(size_t)(mbase + j) * F1P + n0]      = acc0[j];
    xw1[(size_t)(mbase + j) * F1P + n0 + 16] = acc1[j];
  }
}

// ---------------- aggregation layer 1 --------------------------------------
__global__ void gcn_self1(const float* __restrict__ xw1, const float* __restrict__ dinv,
                          float* __restrict__ acc) {
  int idx = blockIdx.x * blockDim.x + threadIdx.x;   // N*64
  if (idx >= NN * F1P) return;
  int i = idx >> 6;
  float d = dinv[i];
  acc[idx] = xw1[idx] * d * d;
}

__global__ void gcn_agg1(const int* __restrict__ ei, int E,
                         const float* __restrict__ xw1, const float* __restrict__ dinv,
                         float* __restrict__ acc) {
  long long idx = (long long)blockIdx.x * blockDim.x + threadIdx.x;  // E*64
  if (idx >= (long long)E * F1P) return;
  int c = (int)(idx & (F1P - 1));
  int e = (int)(idx >> 6);
  if (c >= F1) return;
  int s = ei[e];
  int d = ei[E + e];
  float norm = dinv[s] * dinv[d];
  atomicAdd(&acc[(size_t)d * F1P + c], xw1[(size_t)s * F1P + c] * norm);
}

__global__ void gcn_h1(const float* __restrict__ acc, const float* __restrict__ b1,
                       float* __restrict__ h1) {
  int idx = blockIdx.x * blockDim.x + threadIdx.x;   // N*64
  if (idx >= NN * F1P) return;
  int c = idx & (F1P - 1);
  h1[idx] = (c < F1) ? tanhf(acc[idx] + b1[c]) : 0.0f;
}

// ---------------- layer 2: h1 @ W2, aggregate, tanh ------------------------
__global__ void gcn_l2mm(const float* __restrict__ h1, const float* __restrict__ W2,
                         float* __restrict__ y) {
  int i = blockIdx.x * blockDim.x + threadIdx.x;
  if (i >= NN) return;
  float y0 = 0.0f, y1 = 0.0f;
  const float* row = h1 + (size_t)i * F1P;
#pragma unroll
  for (int k = 0; k < F1; ++k) {
    float v = row[k];
    y0 += v * W2[k * F2 + 0];
    y1 += v * W2[k * F2 + 1];
  }
  y[i * 2 + 0] = y0;
  y[i * 2 + 1] = y1;
}

__global__ void gcn_self2(const float* __restrict__ y, const float* __restrict__ dinv,
                          float* __restrict__ acc) {
  int idx = blockIdx.x * blockDim.x + threadIdx.x;   // N*2
  if (idx >= NN * F2) return;
  int i = idx >> 1;
  float d = dinv[i];
  acc[idx] = y[idx] * d * d;
}

__global__ void gcn_agg2(const int* __restrict__ ei, int E,
                         const float* __restrict__ y, const float* __restrict__ dinv,
                         float* __restrict__ acc) {
  long long idx = (long long)blockIdx.x * blockDim.x + threadIdx.x;  // E*2
  if (idx >= (long long)E * F2) return;
  int c = (int)(idx & 1);
  int e = (int)(idx >> 1);
  int s = ei[e];
  int d = ei[E + e];
  float norm = dinv[s] * dinv[d];
  atomicAdd(&acc[(size_t)d * F2 + c], y[(size_t)s * F2 + c] * norm);
}

// ---------------- final: h = tanh(acc2+b2); out = h @ Wc + bc --------------
__global__ void gcn_final(const float* __restrict__ acc2, const float* __restrict__ b2,
                          const float* __restrict__ Wc, const float* __restrict__ bc,
                          float* __restrict__ out) {
  int i = blockIdx.x * blockDim.x + threadIdx.x;
  if (i >= NN) return;
  float h0 = tanhf(acc2[i * 2 + 0] + b2[0]);
  float h1 = tanhf(acc2[i * 2 + 1] + b2[1]);
  // h is second output, concatenated after out[N,7]
  out[(size_t)NN * NCLS + i * 2 + 0] = h0;
  out[(size_t)NN * NCLS + i * 2 + 1] = h1;
#pragma unroll
  for (int c = 0; c < NCLS; ++c)
    out[(size_t)i * NCLS + c] = h0 * Wc[0 * NCLS + c] + h1 * Wc[1 * NCLS + c] + bc[c];
}

// ---------------------------------------------------------------------------
extern "C" void kernel_launch(void* const* d_in, const int* in_sizes, int n_in,
                              void* d_out, int out_size, void* d_ws, size_t ws_size,
                              hipStream_t stream) {
  const float* x   = (const float*)d_in[0];
  const int*   ei  = (const int*)d_in[1];
  const float* W1  = (const float*)d_in[2];
  const float* b1  = (const float*)d_in[3];
  const float* W2  = (const float*)d_in[4];
  const float* b2  = (const float*)d_in[5];
  const float* Wc  = (const float*)d_in[6];
  const float* bc  = (const float*)d_in[7];
  float*       out = (float*)d_out;
  const int E = in_sizes[1] / 2;

  // workspace layout (floats)
  float*    ws    = (float*)d_ws;
  float*    deg   = ws;                               // N
  float*    dinv  = ws + NN;                          // N
  unsigned* w1pk  = (unsigned*)(ws + 2 * NN);         // NKT*4*32*8 dwords (2MB)
  float*    xw1   = ws + 2 * NN + NKT * 4 * 32 * 8;   // N*64 (reused as h1)
  float*    acc1  = xw1 + (size_t)NN * F1P;           // N*64
  float*    y2    = acc1 + (size_t)NN * F1P;          // N*2
  float*    acc2  = y2 + (size_t)NN * F2;             // N*2

  const int T = 256;
  gcn_deg_init <<<(NN + T - 1) / T, T, 0, stream>>>(deg);
  gcn_deg_edges<<<(E + T - 1) / T, T, 0, stream>>>(ei, E, deg);
  gcn_dinv     <<<(NN + T - 1) / T, T, 0, stream>>>(deg, dinv);

  gcn_pack_w1  <<<(NKT * 4 * 32 + T - 1) / T, T, 0, stream>>>(W1, w1pk);
  gcn_gemm1    <<<NN / MROW, 256, 2 * ABUFB, stream>>>(x, w1pk, xw1);

  gcn_self1    <<<(NN * F1P + T - 1) / T, T, 0, stream>>>(xw1, dinv, acc1);
  gcn_agg1     <<<(int)(((long long)E * F1P + T - 1) / T), T, 0, stream>>>(ei, E, xw1, dinv, acc1);
  gcn_h1       <<<(NN * F1P + T - 1) / T, T, 0, stream>>>(acc1, b1, xw1);  // h1 -> xw1 buf

  gcn_l2mm     <<<(NN + T - 1) / T, T, 0, stream>>>(xw1, W2, y2);
  gcn_self2    <<<(NN * F2 + T - 1) / T, T, 0, stream>>>(y2, dinv, acc2);
  gcn_agg2     <<<(int)(((long long)E * F2 + T - 1) / T), T, 0, stream>>>(ei, E, y2, dinv, acc2);

  gcn_final    <<<(NN + T - 1) / T, T, 0, stream>>>(acc2, b2, Wc, bc, out);
}
